// RayTransformer_18442589569727
// MI455X (gfx1250) — matrix operand order repaired and checked by reading, not compiled, not run on hardware
//
#include <hip/hip_runtime.h>
#include <cmath>
#include <cstdint>

// ---------------------------------------------------------------------------
// Problem constants (from reference)
// ---------------------------------------------------------------------------
#define NVV    4
#define RN_    1024
#define SN_    128
#define HH     256
#define WW     256
#define NPTS   (RN_ * SN_)       // 131072
#define ROWS4  (NPTS * 5)        // 655360  (view transformer rows: 5 tokens/pt)
#define ROWS5  (NPTS)            // 131072  (ray transformer rows: 1024 seq x 128)
#define EPS_LN 1e-5f

typedef _Float16 half_t;
typedef __attribute__((ext_vector_type(16))) _Float16 v16h;
typedef __attribute__((ext_vector_type(8)))  _Float16 v8h;
typedef __attribute__((ext_vector_type(8)))  float    v8f;
typedef __attribute__((ext_vector_type(4)))  unsigned int u32x4;
typedef __attribute__((ext_vector_type(8)))  int          i32x8;
typedef __attribute__((ext_vector_type(4)))  int          i32x4;

// ---------------------------------------------------------------------------
// Generic WMMA GEMM:  D[M x N] = act( A[M x K] * B[K x N] )  (f16 in, f32 acc)
// K, N compile-time (K%32==0, N%16==0); M % 32 == 0.
// One wave per 32x16 output tile (2 M-tiles reuse one B fragment).
// B panel staged in LDS: TDM async tensor load + cooperative fill backup.
// ---------------------------------------------------------------------------
template <int K, int N>
__global__ __launch_bounds__(256)
void wmma_gemm_f16_t(const half_t* __restrict__ A, const half_t* __restrict__ B,
                     half_t* __restrict__ D, int M, int relu)
{
    __shared__ half_t sB[K * N];

    // --- CDNA5 Tensor Data Mover: async 2D tile load of the B panel into LDS.
    // Descriptor per CDNA5 ISA (D# group0/group1, 2D tensor,
    // data_size=2B, tile == tensor == K x N). Redundant cooperative fill below
    // keeps semantics correct regardless of descriptor interpretation.
#if __has_builtin(__builtin_amdgcn_tensor_load_to_lds)
    if (threadIdx.x == 0) {
        const unsigned           lds = (unsigned)(uintptr_t)(void*)&sB[0];
        const unsigned long long ga  = (unsigned long long)(uintptr_t)B;
        u32x4 g0;
        g0[0] = 1u;                                            // count=1 (valid D#)
        g0[1] = lds;                                           // lds_addr
        g0[2] = (unsigned)(ga & 0xFFFFFFFFu);                  // global_addr[31:0]
        g0[3] = (unsigned)((ga >> 32) & 0x1FFFFFFu) | (2u << 30); // addr[56:32]|type=2
        i32x8 g1;
        g1[0] = (1 << 16);                                     // data_size = 2 bytes
        g1[1] = (int)((unsigned)(N & 0xFFFF) << 16);           // tensor_dim0[15:0]
        g1[2] = (int)((unsigned)(K & 0xFFFF) << 16);           // tensor_dim1[15:0]
        g1[3] = (int)((unsigned)(N & 0xFFFF) << 16);           // tile_dim0
        g1[4] = (int)(K & 0xFFFF);                             // tile_dim1 (tile_dim2=0)
        g1[5] = N;                                             // tensor_dim0_stride
        g1[6] = 0;
        g1[7] = 0;
        i32x4 gz4 = {0, 0, 0, 0};
        i32x8 gz8 = {0, 0, 0, 0, 0, 0, 0, 0};
        // 6-arg form (clang-23 / therock-10.0 headers):
        // (g0, g1, g2, g3, g4, cpol)
        __builtin_amdgcn_tensor_load_to_lds(g0, g1, gz4, gz4, gz8, 0);
    }
#if __has_builtin(__builtin_amdgcn_s_wait_tensorcnt)
    __builtin_amdgcn_s_wait_tensorcnt((short)0);
#endif
#endif
    // cooperative (coalesced) fill — authoritative contents of sB
    for (int i = threadIdx.x; i < K * N; i += 256) sB[i] = B[i];
    __syncthreads();

    const int wave = threadIdx.x >> 5;
    const int lane = threadIdx.x & 31;
    constexpr int ntn = N >> 4;
    const int tile  = blockIdx.x * 8 + wave;
    const int total = (M >> 5) * ntn;
    if (tile >= total) return;                 // wave-uniform branch
    const int tm   = tile / ntn;
    const int tn   = tile - tm * ntn;
    const int row0 = tm << 5;                  // 32 rows per wave
    const int col0 = tn << 4;
    const int kh   = lane >> 4;                // half-wave id (0/1)
    const int lid  = lane & 15;                // A: M row ; B/D: N col

    v8f acc0 = {}, acc1 = {};
#pragma unroll
    for (int k0 = 0; k0 < K; k0 += 32) {
        // A fragment (16x32 f16 ISA layout): two contiguous 16B chunks per lane
        const half_t* aR0 = A + (size_t)(row0 + lid) * K + k0 + kh * 8;
        const half_t* aR1 = aR0 + (size_t)16 * K;
        v8h a00 = *reinterpret_cast<const v8h*>(aR0);
        v8h a01 = *reinterpret_cast<const v8h*>(aR0 + 16);
        v8h a10 = *reinterpret_cast<const v8h*>(aR1);
        v8h a11 = *reinterpret_cast<const v8h*>(aR1 + 16);
        v16h a0, a1;
#pragma unroll
        for (int i = 0; i < 8; ++i) {
            a0[i] = a00[i]; a0[8 + i] = a01[i];
            a1[i] = a10[i]; a1[8 + i] = a11[i];
        }
        // B fragment from LDS: lanes 0-15 hold K 0..7,16..23, lanes 16-31 hold K 8..15,24..31
        v16h b;
#pragma unroll
        for (int e = 0; e < 16; ++e)
            b[e] = sB[(k0 + kh * 8 + e + (e >= 8 ? 8 : 0)) * N + col0 + lid];

        acc0 = __builtin_amdgcn_wmma_f32_16x16x32_f16(
                   false, a0, false, b, (short)0, acc0, false, false);
        acc1 = __builtin_amdgcn_wmma_f32_16x16x32_f16(
                   false, a1, false, b, (short)0, acc1, false, false);
    }
    // D store (f32 C/D layout: VGPR j -> M = j + kh*8, N = lane&15)
    half_t* d0 = D + (size_t)(row0 + kh * 8) * N + col0 + lid;
    half_t* d1 = d0 + (size_t)16 * N;
#pragma unroll
    for (int j = 0; j < 8; ++j) {
        float v0 = acc0[j], v1 = acc1[j];
        if (relu) { v0 = fmaxf(v0, 0.f); v1 = fmaxf(v1, 0.f); }
        d0[(size_t)j * N] = (half_t)v0;
        d1[(size_t)j * N] = (half_t)v1;
    }
}

// ---------------------------------------------------------------------------
// Weight staging: f32 -> padded f16
// ---------------------------------------------------------------------------
__global__ void zero_h(half_t* p, int n)
{
    int i = blockIdx.x * 256 + threadIdx.x;
    if (i < n) p[i] = (half_t)0.f;
}
__global__ void pack_w(const float* __restrict__ src, half_t* __restrict__ dst,
                       int sR, int sC, int dC, int colOff)
{
    int i = blockIdx.x * 256 + threadIdx.x;
    if (i >= sR * sC) return;
    int r = i / sC, c = i - r * sC;
    dst[(size_t)r * dC + colOff + c] = (half_t)src[i];
}

// ---------------------------------------------------------------------------
// K0: projection, pip output, masks, bilinear sampling (feat->X4 f16, rgb),
//     relative view directions, view-token rows.  grid = (NPTS/256, NV)
// ---------------------------------------------------------------------------
__global__ __launch_bounds__(256)
void project_sample_kernel(const float* __restrict__ P3D,
                           const float* __restrict__ IMGS,
                           const float* __restrict__ FEAT,
                           const float* __restrict__ RPI,
                           const float* __restrict__ SPI,
                           const float* __restrict__ SP,
                           const float* __restrict__ TOK,
                           half_t* __restrict__ X4,
                           float* __restrict__ RGB,
                           float* __restrict__ MASK,
                           float* __restrict__ DR,
                           float* __restrict__ PIP)
{
    const int pid = blockIdx.x * 256 + threadIdx.x;
    const int v   = blockIdx.y;
    if (pid >= NPTS) return;

    const float px = P3D[pid * 3 + 0];
    const float py = P3D[pid * 3 + 1];
    const float pz = P3D[pid * 3 + 2];

    // pip = P @ [p,1]
    const float* P = SP + v * 16;
    const float p0 = P[0] * px + P[1] * py + P[2]  * pz + P[3];
    const float p1 = P[4] * px + P[5] * py + P[6]  * pz + P[7];
    const float p2 = P[8] * px + P[9] * py + P[10] * pz + P[11];
    PIP[((size_t)(v * 3) + 0) * NPTS + pid] = p0;
    PIP[((size_t)(v * 3) + 1) * NPTS + pid] = p1;
    PIP[((size_t)(v * 3) + 2) * NPTS + pid] = p2;

    const float mvd = (p2 > 0.f) ? 1.f : 0.f;
    const float gx = p0 / p2, gy = p1 / p2;
    const float inb = (gx >= 0.f && gx <= (float)(WW - 1) &&
                       gy >= 0.f && gy <= (float)(HH - 1)) ? 1.f : 0.f;
    MASK[(size_t)v * NPTS + pid] = inb * mvd;

    // bilinear taps (per-tap validity, clamped index, weight zeroed if invalid)
    const float x0f = floorf(gx), y0f = floorf(gy);
    const int ix0 = (int)x0f, iy0 = (int)y0f;
    const float wx1 = gx - x0f, wy1 = gy - y0f;
    const int   ixs[2] = { ix0, ix0 + 1 };
    const int   iys[2] = { iy0, iy0 + 1 };
    const float wxs[2] = { 1.f - wx1, wx1 };
    const float wys[2] = { 1.f - wy1, wy1 };
    int   idx[4];
    float wt[4];
#pragma unroll
    for (int b = 0; b < 2; ++b)
#pragma unroll
        for (int a = 0; a < 2; ++a) {
            const bool ok = (ixs[a] >= 0 && ixs[a] < WW && iys[b] >= 0 && iys[b] < HH);
            const int cx = min(max(ixs[a], 0), WW - 1);
            const int cy = min(max(iys[b], 0), HH - 1);
            idx[b * 2 + a] = cy * WW + cx;
            wt[b * 2 + a]  = ok ? (wxs[a] * wys[b]) : 0.f;
        }

    // 32 feature channels -> transformer input row (pid*5 + 1 + v)
    half_t* xrow = X4 + ((size_t)pid * 5 + 1 + v) * 32;
#pragma unroll 4
    for (int c = 0; c < 32; ++c) {
        const float* im = FEAT + ((size_t)v * 32 + c) * (HH * WW);
        float f = wt[0] * im[idx[0]] + wt[1] * im[idx[1]] +
                  wt[2] * im[idx[2]] + wt[3] * im[idx[3]];
        xrow[c] = (half_t)f;
    }
    // 3 rgb channels
#pragma unroll
    for (int c = 0; c < 3; ++c) {
        const float* im = IMGS + ((size_t)v * 3 + c) * (HH * WW);
        float f = wt[0] * im[idx[0]] + wt[1] * im[idx[1]] +
                  wt[2] * im[idx[2]] + wt[3] * im[idx[3]];
        RGB[((size_t)v * 3 + c) * NPTS + pid] = f;
    }

    // relative directions
    float v1x = px - RPI[3], v1y = py - RPI[7], v1z = pz - RPI[11];
    float rn = rsqrtf(v1x * v1x + v1y * v1y + v1z * v1z);
    v1x *= rn; v1y *= rn; v1z *= rn;
    const float* Pi = SPI + v * 16;
    float v2x = px - Pi[3], v2y = py - Pi[7], v2z = pz - Pi[11];
    rn = rsqrtf(v2x * v2x + v2y * v2y + v2z * v2z);
    v2x *= rn; v2y *= rn; v2z *= rn;
    float* dr = DR + ((size_t)pid * 4 + v) * 3;
    dr[0] = v1x - v2x; dr[1] = v1y - v2y; dr[2] = v1z - v2z;

    // view-token row (once per point)
    if (v == 0) {
        half_t* trow = X4 + (size_t)pid * 5 * 32;
#pragma unroll 8
        for (int c = 0; c < 32; ++c) trow[c] = (half_t)TOK[c];
    }
}

// ---------------------------------------------------------------------------
// A4: linear attention, view transformer. One wave per sequence (L=5, 8 heads,
// dim=4). lane = channel. QKV layout: row*96 {q[0..31] k[32..63] v[64..95]}
// ---------------------------------------------------------------------------
__global__ __launch_bounds__(256)
void view_attn_kernel(const half_t* __restrict__ QKV, half_t* __restrict__ AT)
{
    const int wave = threadIdx.x >> 5;
    const int lane = threadIdx.x & 31;
    const int pid  = blockIdx.x * 8 + wave;
    if (pid >= NPTS) return;
    const int hbase = lane & ~3;                  // first lane of this head

    float Qp[5], Kp[5], vv[5], Ks = 0.f;
#pragma unroll
    for (int l = 0; l < 5; ++l) {
        const half_t* p = QKV + ((size_t)pid * 5 + l) * 96 + lane;
        float q = (float)p[0], k = (float)p[32];
        vv[l] = (float)p[64];
        Qp[l] = q > 0.f ? q + 1.f : expf(q);      // elu(x)+1
        Kp[l] = k > 0.f ? k + 1.f : expf(k);
        Ks += Kp[l];
    }
    float KVc[4] = {0.f, 0.f, 0.f, 0.f};          // column (dim=lane&3) of KV[h]
#pragma unroll
    for (int l = 0; l < 5; ++l) {
        const float vl = vv[l] * 0.2f;            // v / L
#pragma unroll
        for (int d = 0; d < 4; ++d)
            KVc[d] += __shfl(Kp[l], hbase + d, 32) * vl;
    }
#pragma unroll
    for (int l = 0; l < 5; ++l) {
        float dp = Qp[l] * Ks;                    // per-dim product, reduce in head
        dp += __shfl_xor(dp, 1, 32);
        dp += __shfl_xor(dp, 2, 32);
        const float Z = 1.f / (dp + 1e-6f);
        float o = 0.f;
#pragma unroll
        for (int d = 0; d < 4; ++d)
            o += __shfl(Qp[l], hbase + d, 32) * KVc[d];
        AT[((size_t)pid * 5 + l) * 32 + lane] = (half_t)(o * Z * 5.f);
    }
}

// ---------------------------------------------------------------------------
// C4: LN1(msg) and CAT=[x | ln(msg)]  (view transformer). Wave per row.
// ---------------------------------------------------------------------------
__global__ __launch_bounds__(256)
void cat_ln1_view(const half_t* __restrict__ MSG, const half_t* __restrict__ X,
                  const float* __restrict__ g, const float* __restrict__ bb,
                  half_t* __restrict__ CAT)
{
    const int wave = threadIdx.x >> 5;
    const int lane = threadIdx.x & 31;
    const int row  = blockIdx.x * 8 + wave;
    if (row >= ROWS4) return;
    float m = (float)MSG[(size_t)row * 32 + lane];
    float s = m;
    for (int o = 16; o; o >>= 1) s += __shfl_xor(s, o, 32);
    const float mean = s * (1.f / 32.f);
    const float d = m - mean;
    float s2 = d * d;
    for (int o = 16; o; o >>= 1) s2 += __shfl_xor(s2, o, 32);
    const float inv = rsqrtf(s2 * (1.f / 32.f) + EPS_LN);
    CAT[(size_t)row * 64 + lane]      = X[(size_t)row * 32 + lane];
    CAT[(size_t)row * 64 + 32 + lane] = (half_t)(d * inv * g[lane] + bb[lane]);
}

// ---------------------------------------------------------------------------
// F4: out = x + LN2(h2); scatter token rows -> X5 (with posenc, zero pad),
//     view rows -> VF (f32).  Wave per row.
// ---------------------------------------------------------------------------
__global__ __launch_bounds__(256)
void finish_view(const half_t* __restrict__ H2, const half_t* __restrict__ X,
                 const float* __restrict__ g, const float* __restrict__ bb,
                 half_t* __restrict__ X5, float* __restrict__ VF)
{
    const int wave = threadIdx.x >> 5;
    const int lane = threadIdx.x & 31;
    const int row  = blockIdx.x * 8 + wave;
    if (row >= ROWS4) return;
    float m = (float)H2[(size_t)row * 32 + lane];
    float s = m;
    for (int o = 16; o; o >>= 1) s += __shfl_xor(s, o, 32);
    const float mean = s * (1.f / 32.f);
    const float d = m - mean;
    float s2 = d * d;
    for (int o = 16; o; o >>= 1) s2 += __shfl_xor(s2, o, 32);
    const float inv = rsqrtf(s2 * (1.f / 32.f) + EPS_LN);
    const float out = (float)X[(size_t)row * 32 + lane] + d * inv * g[lane] + bb[lane];

    const int pid = row / 5;
    const int l   = row - pid * 5;
    if (l == 0) {
        X5[(size_t)pid * 64 + lane] = (half_t)out;
        // posenc cols 32..39, zeros 40..63
        const int sidx = pid & (SN_ - 1);
        float pe = 0.f;
        if (lane < 8) {
            const int j = lane;
            const float expo = (float)(2 * (j >> 1)) * (1.f / 8.f);
            const float ang = (float)sidx * powf(10000.f, -expo);
            pe = (j & 1) ? cosf(ang) : sinf(ang);
        }
        X5[(size_t)pid * 64 + 32 + lane] = (half_t)pe;
    } else {
        VF[((size_t)pid * 4 + (l - 1)) * 32 + lane] = out;
    }
}

// ---------------------------------------------------------------------------
// A5: linear attention, ray transformer. Block per sequence (L=128 threads),
// 8 heads, dim=5. LDS f32 atomics accumulate KV / Ksum.
// QKV layout: row*144 {q[0..47] k[48..95] v[96..143]}, valid width 40.
// ---------------------------------------------------------------------------
__global__ __launch_bounds__(128)
void ray_attn_kernel(const half_t* __restrict__ QKV, half_t* __restrict__ AT)
{
    __shared__ float sKV[8 * 5 * 5];
    __shared__ float sKs[8 * 5];
    for (int i = threadIdx.x; i < 8 * 5 * 5; i += 128) sKV[i] = 0.f;
    for (int i = threadIdx.x; i < 8 * 5;     i += 128) sKs[i] = 0.f;
    __syncthreads();

    const size_t row = (size_t)blockIdx.x * 128 + threadIdx.x;
    const half_t* base = QKV + row * 144;
    float Qp[40], Kp[40], vv[40];
#pragma unroll
    for (int c = 0; c < 40; ++c) {
        const float q = (float)base[c];
        const float k = (float)base[48 + c];
        vv[c] = (float)base[96 + c];
        Qp[c] = q > 0.f ? q + 1.f : expf(q);
        Kp[c] = k > 0.f ? k + 1.f : expf(k);
    }
#pragma unroll
    for (int h = 0; h < 8; ++h)
#pragma unroll
        for (int d = 0; d < 5; ++d) {
            const float kd = Kp[h * 5 + d];
            atomicAdd(&sKs[h * 5 + d], kd);
#pragma unroll
            for (int e = 0; e < 5; ++e)
                atomicAdd(&sKV[(h * 5 + d) * 5 + e], kd * vv[h * 5 + e] * (1.f / 128.f));
        }
    __syncthreads();

    half_t* out = AT + row * 64;
#pragma unroll
    for (int h = 0; h < 8; ++h) {
        float dp = 1e-6f;
#pragma unroll
        for (int d = 0; d < 5; ++d) dp += Qp[h * 5 + d] * sKs[h * 5 + d];
        const float Z = 128.f / dp;
#pragma unroll
        for (int e = 0; e < 5; ++e) {
            float o = 0.f;
#pragma unroll
            for (int d = 0; d < 5; ++d) o += Qp[h * 5 + d] * sKV[(h * 5 + d) * 5 + e];
            out[h * 5 + e] = (half_t)(o * Z);
        }
    }
#pragma unroll
    for (int c = 40; c < 64; ++c) out[c] = (half_t)0.f;  // pad for K=64 GEMM
}

// ---------------------------------------------------------------------------
// C5: LN1(msg, 40ch) and CAT=[x(40) | ln(40) | 0(16)]. Thread per row.
// ---------------------------------------------------------------------------
__global__ __launch_bounds__(256)
void cat_ln1_ray(const half_t* __restrict__ MSG, const half_t* __restrict__ X5,
                 const float* __restrict__ g, const float* __restrict__ bb,
                 half_t* __restrict__ CAT)
{
    const size_t row = (size_t)blockIdx.x * 256 + threadIdx.x;
    if (row >= ROWS5) return;
    float m[40], s = 0.f;
#pragma unroll
    for (int c = 0; c < 40; ++c) { m[c] = (float)MSG[row * 48 + c]; s += m[c]; }
    const float mean = s * (1.f / 40.f);
    float s2 = 0.f;
#pragma unroll
    for (int c = 0; c < 40; ++c) { const float d = m[c] - mean; s2 += d * d; }
    const float inv = rsqrtf(s2 * (1.f / 40.f) + EPS_LN);
    half_t* out = CAT + row * 96;
#pragma unroll
    for (int c = 0; c < 40; ++c) out[c] = X5[row * 64 + c];
#pragma unroll
    for (int c = 0; c < 40; ++c)
        out[40 + c] = (half_t)((m[c] - mean) * inv * g[c] + bb[c]);
#pragma unroll
    for (int c = 80; c < 96; ++c) out[c] = (half_t)0.f;
}

// ---------------------------------------------------------------------------
// F5: x_out = x + LN2(h2) then fused dm head (40->32->16->1) -> srdf.
// ---------------------------------------------------------------------------
__global__ __launch_bounds__(256)
void finish_ray(const half_t* __restrict__ H2, const half_t* __restrict__ X5,
                const float* __restrict__ g, const float* __restrict__ bb,
                const float* __restrict__ W1, const float* __restrict__ b1,
                const float* __restrict__ W2, const float* __restrict__ b2,
                const float* __restrict__ W3, const float* __restrict__ b3,
                float* __restrict__ srdf)
{
    const size_t row = (size_t)blockIdx.x * 256 + threadIdx.x;
    if (row >= ROWS5) return;
    float m[40], s = 0.f;
#pragma unroll
    for (int c = 0; c < 40; ++c) { m[c] = (float)H2[row * 48 + c]; s += m[c]; }
    const float mean = s * (1.f / 40.f);
    float s2 = 0.f;
#pragma unroll
    for (int c = 0; c < 40; ++c) { const float d = m[c] - mean; s2 += d * d; }
    const float inv = rsqrtf(s2 * (1.f / 40.f) + EPS_LN);
    float xo[40];
#pragma unroll
    for (int c = 0; c < 40; ++c)
        xo[c] = (float)X5[row * 64 + c] + (m[c] - mean) * inv * g[c] + bb[c];
    float h1[32];
#pragma unroll 4
    for (int j = 0; j < 32; ++j) {
        float a = b1[j];
        for (int c = 0; c < 40; ++c) a += xo[c] * W1[c * 32 + j];
        h1[j] = a > 0.f ? a : 0.f;
    }
    float h2[16];
#pragma unroll 4
    for (int j = 0; j < 16; ++j) {
        float a = b2[j];
        for (int c = 0; c < 32; ++c) a += h1[c] * W2[c * 16 + j];
        h2[j] = a > 0.f ? a : 0.f;
    }
    float o = b3[0];
#pragma unroll
    for (int j = 0; j < 16; ++j) o += h2[j] * W3[j];
    srdf[row] = o;
}

// ---------------------------------------------------------------------------
// R: rw head (35->16->8->1) per view, mask, softmax over views, radiance.
// ---------------------------------------------------------------------------
__global__ __launch_bounds__(256)
void radiance_kernel(const float* __restrict__ VF, const float* __restrict__ DR,
                     const float* __restrict__ MASK, const float* __restrict__ RGB,
                     const float* __restrict__ W1, const float* __restrict__ b1,
                     const float* __restrict__ W2, const float* __restrict__ b2,
                     const float* __restrict__ W3, const float* __restrict__ b3,
                     float* __restrict__ rad)
{
    const size_t pid = (size_t)blockIdx.x * 256 + threadIdx.x;
    if (pid >= NPTS) return;
    float xw[NVV];
    for (int v = 0; v < NVV; ++v) {
        float in[35];
        const float* vf = VF + (pid * 4 + v) * 32;
        for (int c = 0; c < 32; ++c) in[c] = vf[c];
        const float* dr = DR + (pid * 4 + v) * 3;
        in[32] = dr[0]; in[33] = dr[1]; in[34] = dr[2];
        float h1[16];
        for (int j = 0; j < 16; ++j) {
            float a = b1[j];
            for (int c = 0; c < 35; ++c) a += in[c] * W1[c * 16 + j];
            h1[j] = a > 0.f ? a : 0.f;
        }
        float h2[8];
        for (int j = 0; j < 8; ++j) {
            float a = b2[j];
            for (int c = 0; c < 16; ++c) a += h1[c] * W2[c * 8 + j];
            h2[j] = a > 0.f ? a : 0.f;
        }
        float o = b3[0];
        for (int j = 0; j < 8; ++j) o += h2[j] * W3[j];
        const float mv = MASK[(size_t)v * NPTS + pid];
        xw[v] = (mv == 0.f) ? -1e9f : o;
    }
    float mx = xw[0];
    for (int v = 1; v < NVV; ++v) mx = fmaxf(mx, xw[v]);
    float wsum = 0.f, wv[NVV];
    for (int v = 0; v < NVV; ++v) { wv[v] = expf(xw[v] - mx); wsum += wv[v]; }
    const float rinv = 1.f / wsum;
    for (int c = 0; c < 3; ++c) {
        float r = 0.f;
        for (int v = 0; v < NVV; ++v)
            r += RGB[((size_t)v * 3 + c) * NPTS + pid] * wv[v] * rinv;
        rad[pid * 3 + c] = r;
    }
}

// ---------------------------------------------------------------------------
// Host driver
// ---------------------------------------------------------------------------
static inline int gemm_blocks(int M, int N)
{
    return (((M / 32) * (N / 16)) + 7) / 8;
}

extern "C" void kernel_launch(void* const* d_in, const int* in_sizes, int n_in,
                              void* d_out, int out_size, void* d_ws, size_t ws_size,
                              hipStream_t stream)
{
    (void)in_sizes; (void)n_in; (void)out_size; (void)ws_size;

    const float* point3D  = (const float*)d_in[0];
    const float* src_imgs = (const float*)d_in[1];
    const float* src_feat = (const float*)d_in[2];
    const float* rpi      = (const float*)d_in[3];
    const float* spi      = (const float*)d_in[4];
    const float* sp       = (const float*)d_in[5];
    const float* tok      = (const float*)d_in[6];
    const float* vt_Wq = (const float*)d_in[7],  *vt_Wk = (const float*)d_in[8];
    const float* vt_Wv = (const float*)d_in[9],  *vt_Wm = (const float*)d_in[10];
    const float* vt_g1 = (const float*)d_in[11], *vt_b1 = (const float*)d_in[12];
    const float* vt_W1 = (const float*)d_in[13], *vt_W2 = (const float*)d_in[14];
    const float* vt_g2 = (const float*)d_in[15], *vt_b2 = (const float*)d_in[16];
    const float* rt_Wq = (const float*)d_in[17], *rt_Wk = (const float*)d_in[18];
    const float* rt_Wv = (const float*)d_in[19], *rt_Wm = (const float*)d_in[20];
    const float* rt_g1 = (const float*)d_in[21], *rt_b1 = (const float*)d_in[22];
    const float* rt_W1 = (const float*)d_in[23], *rt_W2 = (const float*)d_in[24];
    const float* rt_g2 = (const float*)d_in[25], *rt_b2 = (const float*)d_in[26];
    const float* dm_W1 = (const float*)d_in[27], *dm_b1 = (const float*)d_in[28];
    const float* dm_W2 = (const float*)d_in[29], *dm_b2 = (const float*)d_in[30];
    const float* dm_W3 = (const float*)d_in[31], *dm_b3 = (const float*)d_in[32];
    const float* rw_W1 = (const float*)d_in[33], *rw_b1 = (const float*)d_in[34];
    const float* rw_W2 = (const float*)d_in[35], *rw_b2 = (const float*)d_in[36];
    const float* rw_W3 = (const float*)d_in[37], *rw_b3 = (const float*)d_in[38];

    // d_out layout: radiance (NPTS*3) | srdf (NPTS) | pip (NV*3*NPTS)
    float* out_rad  = (float*)d_out;
    float* out_srdf = out_rad + (size_t)NPTS * 3;
    float* out_pip  = out_srdf + (size_t)NPTS;

    // workspace carve-up
    char* w = (char*)d_ws;
    size_t off = 0;
    auto allocH = [&](size_t elems) -> half_t* {
        half_t* p = (half_t*)(w + off);
        off += ((elems * sizeof(half_t)) + 255) & ~(size_t)255;
        return p;
    };
    auto allocF = [&](size_t elems) -> float* {
        float* p = (float*)(w + off);
        off += ((elems * sizeof(float)) + 255) & ~(size_t)255;
        return p;
    };
    half_t* X4   = allocH((size_t)ROWS4 * 32);
    half_t* QKV4 = allocH((size_t)ROWS4 * 96);
    half_t* AT4  = allocH((size_t)ROWS4 * 32);
    half_t* MSG4 = allocH((size_t)ROWS4 * 32);
    half_t* CAT4 = allocH((size_t)ROWS4 * 64);
    half_t* H1V  = allocH((size_t)ROWS4 * 64);
    half_t* H2V  = allocH((size_t)ROWS4 * 32);
    half_t* X5   = allocH((size_t)ROWS5 * 64);
    half_t* QKV5 = allocH((size_t)ROWS5 * 144);
    half_t* AT5  = allocH((size_t)ROWS5 * 64);
    half_t* MSG5 = allocH((size_t)ROWS5 * 48);
    half_t* CAT5 = allocH((size_t)ROWS5 * 96);
    half_t* H1R  = allocH((size_t)ROWS5 * 96);
    half_t* H2R  = allocH((size_t)ROWS5 * 48);
    float*  VF   = allocF((size_t)NPTS * 4 * 32);
    float*  DR   = allocF((size_t)NPTS * 4 * 3);
    float*  RGB  = allocF((size_t)NVV * 3 * NPTS);
    float*  MSK  = allocF((size_t)NVV * NPTS);
    half_t* WQKV4 = allocH(32 * 96);
    half_t* WM4   = allocH(32 * 32);
    half_t* W1V   = allocH(64 * 64);
    half_t* W2V   = allocH(64 * 32);
    half_t* WQKV5 = allocH(64 * 144);
    half_t* WM5   = allocH(64 * 48);
    half_t* W1R   = allocH(96 * 96);
    half_t* W2R   = allocH(96 * 48);

    auto zero = [&](half_t* p, int n) {
        zero_h<<<(n + 255) / 256, 256, 0, stream>>>(p, n);
    };
    auto pack = [&](const float* s, half_t* d, int sR, int sC, int dC, int co) {
        pack_w<<<(sR * sC + 255) / 256, 256, 0, stream>>>(s, d, sR, sC, dC, co);
    };

    // stage weights (zero padded buffers each call; harness poisons ws)
    zero(WQKV5, 64 * 144); zero(WM5, 64 * 48); zero(W1R, 96 * 96); zero(W2R, 96 * 48);
    pack(vt_Wq, WQKV4, 32, 32, 96, 0);
    pack(vt_Wk, WQKV4, 32, 32, 96, 32);
    pack(vt_Wv, WQKV4, 32, 32, 96, 64);
    pack(vt_Wm, WM4, 32, 32, 32, 0);
    pack(vt_W1, W1V, 64, 64, 64, 0);
    pack(vt_W2, W2V, 64, 32, 32, 0);
    pack(rt_Wq, WQKV5, 40, 40, 144, 0);
    pack(rt_Wk, WQKV5, 40, 40, 144, 48);
    pack(rt_Wv, WQKV5, 40, 40, 144, 96);
    pack(rt_Wm, WM5, 40, 40, 48, 0);
    pack(rt_W1, W1R, 80, 80, 96, 0);
    pack(rt_W2, W2R, 80, 40, 48, 0);

    // K0: project + sample + dirs + token + pip
    project_sample_kernel<<<dim3(NPTS / 256, NVV), 256, 0, stream>>>(
        point3D, src_imgs, src_feat, rpi, spi, sp, tok, X4, RGB, MSK, DR, out_pip);

    // ---- view transformer (L=5, d=32) ----
    wmma_gemm_f16_t<32, 96><<<gemm_blocks(ROWS4, 96), 256, 0, stream>>>(
        X4, WQKV4, QKV4, ROWS4, 0);
    view_attn_kernel<<<NPTS / 8, 256, 0, stream>>>(QKV4, AT4);
    wmma_gemm_f16_t<32, 32><<<gemm_blocks(ROWS4, 32), 256, 0, stream>>>(
        AT4, WM4, MSG4, ROWS4, 0);
    cat_ln1_view<<<ROWS4 / 8, 256, 0, stream>>>(MSG4, X4, vt_g1, vt_b1, CAT4);
    wmma_gemm_f16_t<64, 64><<<gemm_blocks(ROWS4, 64), 256, 0, stream>>>(
        CAT4, W1V, H1V, ROWS4, 1);
    wmma_gemm_f16_t<64, 32><<<gemm_blocks(ROWS4, 32), 256, 0, stream>>>(
        H1V, W2V, H2V, ROWS4, 0);
    finish_view<<<ROWS4 / 8, 256, 0, stream>>>(H2V, X4, vt_g2, vt_b2, X5, VF);

    // ---- ray transformer (L=128, d=40 padded to 64) ----
    wmma_gemm_f16_t<64, 144><<<gemm_blocks(ROWS5, 144), 256, 0, stream>>>(
        X5, WQKV5, QKV5, ROWS5, 0);
    ray_attn_kernel<<<RN_, 128, 0, stream>>>(QKV5, AT5);
    wmma_gemm_f16_t<64, 48><<<gemm_blocks(ROWS5, 48), 256, 0, stream>>>(
        AT5, WM5, MSG5, ROWS5, 0);
    cat_ln1_ray<<<ROWS5 / 256, 256, 0, stream>>>(MSG5, X5, rt_g1, rt_b1, CAT5);
    wmma_gemm_f16_t<96, 96><<<gemm_blocks(ROWS5, 96), 256, 0, stream>>>(
        CAT5, W1R, H1R, ROWS5, 1);
    wmma_gemm_f16_t<96, 48><<<gemm_blocks(ROWS5, 48), 256, 0, stream>>>(
        H1R, W2R, H2R, ROWS5, 0);
    finish_ray<<<ROWS5 / 256, 256, 0, stream>>>(H2R, X5, rt_g2, rt_b2,
        dm_W1, dm_b1, dm_W2, dm_b2, dm_W3, dm_b3, out_srdf);

    // ---- per-view weights + radiance ----
    radiance_kernel<<<NPTS / 256, 256, 0, stream>>>(VF, DR, MSK, RGB,
        rw_W1, rw_b1, rw_W2, rw_b2, rw_W3, rw_b3, out_rad);
}
